// LinearAttention2d_87625922773585
// MI455X (gfx1250) — matrix operand order repaired and checked by reading, not compiled, not run on hardware
//
#include <hip/hip_runtime.h>

typedef __attribute__((ext_vector_type(16))) __bf16 v16bf;
typedef __attribute__((ext_vector_type(8)))  float  v8f;

union Frag { v16bf v; unsigned int u[8]; };

__device__ __forceinline__ unsigned short f32_to_bf16(float f) {
    unsigned int u = __float_as_uint(f);
    u = (u + 0x7FFFu + ((u >> 16) & 1u)) >> 16;   // round-to-nearest-even
    return (unsigned short)u;
}
__device__ __forceinline__ float bf16_to_f32(unsigned short h) {
    return __uint_as_float(((unsigned int)h) << 16);
}

// ---------------------------------------------------------------------------
// Generic tiled GEMM: C[M,N] = A[M,K] * B[K,N] (+bias), bf16 WMMA, f32 accum.
// A row-major [m][k] (f32 or bf16), B row-major [k][n] (f32 or bf16).
// Per-block batch offset: off = (z/hdiv)*S1 + (z%hdiv)*S2  (elements).
// Tile: TM x 128, K-stage 64 (2 WMMA K-steps per stage).
// 256 threads = 8 waves (2x4 wave grid). Vectorized b128 staging + prefetch.
// ---------------------------------------------------------------------------
template <int TM, bool A_F32, bool B_F32, bool OUT_F32, bool BIAS>
__global__ __launch_bounds__(256) void gemm_bf16_kernel(
    const void* __restrict__ Ap, int lda, long long a_s1, long long a_s2,
    const void* __restrict__ Bp, int ldb, long long b_s1, long long b_s2,
    void* __restrict__ Cp,       int ldc, long long c_s1, long long c_s2,
    const float* __restrict__ bias, int K, int hdiv)
{
    constexpr int TN = 128, TK = 64, MW = 2, NW = 4;
    constexpr int FM = TM / (16 * MW);   // frag rows per wave
    constexpr int FN = TN / (16 * NW);   // frag cols per wave (=2)
    constexpr int BPAD = TK + 4;         // 68: even pad -> 4B-aligned pairs
    constexpr int AV = A_F32 ? 4 : 8;    // elems per vector load (16B)
    constexpr int BV = B_F32 ? 4 : 8;

    __shared__ unsigned short lA[TM * TK];
    __shared__ unsigned short lB[TN * BPAD];

    const int z = blockIdx.z;
    const long long aoff = (long long)(z / hdiv) * a_s1 + (long long)(z % hdiv) * a_s2;
    const long long boff = (long long)(z / hdiv) * b_s1 + (long long)(z % hdiv) * b_s2;
    const long long coff = (long long)(z / hdiv) * c_s1 + (long long)(z % hdiv) * c_s2;

    const int m0 = blockIdx.y * TM;
    const int n0 = blockIdx.x * TN;
    const int t    = threadIdx.x;
    const int wave = t >> 5;
    const int lane = t & 31;
    const int l    = lane & 15;
    const int hi   = lane >> 4;
    const int wm   = wave % MW;
    const int wn   = wave / MW;

    const float*          Af  = (const float*)Ap;
    const unsigned short* Abf = (const unsigned short*)Ap;
    const float*          Bf  = (const float*)Bp;
    const unsigned short* Bbf = (const unsigned short*)Bp;

    v8f acc[FM][FN] = {};

    for (int kb = 0; kb < K; kb += TK) {
        // ---- stage A tile [m][k] as bf16 (b128 global loads) ----
        for (int i = t; i < TM * TK / AV; i += 256) {
            int base = i * AV;
            int m = base / TK, k = base % TK;
            long long g = aoff + (long long)(m0 + m) * lda + (kb + k);
            if (A_F32) {
                const float4 fv = *(const float4*)&Af[g];
                uint2 p;
                p.x = (unsigned int)f32_to_bf16(fv.x) | ((unsigned int)f32_to_bf16(fv.y) << 16);
                p.y = (unsigned int)f32_to_bf16(fv.z) | ((unsigned int)f32_to_bf16(fv.w) << 16);
                *(uint2*)&lA[m * TK + k] = p;
            } else {
                *(uint4*)&lA[m * TK + k] = *(const uint4*)&Abf[g];
            }
        }
        // ---- stage B tile transposed [n][k] as bf16 (b128 global loads) ----
        for (int i = t; i < TK * TN / BV; i += 256) {
            int base = i * BV;
            int n = base % TN, k = base / TN;            // coalesced over n
            long long g = boff + (long long)(kb + k) * ldb + (n0 + n);
            if (B_F32) {
                const float4 fv = *(const float4*)&Bf[g];
                lB[(n + 0) * BPAD + k] = f32_to_bf16(fv.x);
                lB[(n + 1) * BPAD + k] = f32_to_bf16(fv.y);
                lB[(n + 2) * BPAD + k] = f32_to_bf16(fv.z);
                lB[(n + 3) * BPAD + k] = f32_to_bf16(fv.w);
            } else {
                const uint4 u = *(const uint4*)&Bbf[g];
                lB[(n + 0) * BPAD + k] = (unsigned short)u.x;
                lB[(n + 1) * BPAD + k] = (unsigned short)(u.x >> 16);
                lB[(n + 2) * BPAD + k] = (unsigned short)u.y;
                lB[(n + 3) * BPAD + k] = (unsigned short)(u.y >> 16);
                lB[(n + 4) * BPAD + k] = (unsigned short)u.z;
                lB[(n + 5) * BPAD + k] = (unsigned short)(u.z >> 16);
                lB[(n + 6) * BPAD + k] = (unsigned short)u.w;
                lB[(n + 7) * BPAD + k] = (unsigned short)(u.w >> 16);
            }
        }
        // ---- prefetch next K-stage while compute runs ----
        if (kb + TK < K) {
            long long ga = aoff + (long long)(m0 + (t * AV) / TK) * lda + (kb + TK + (t * AV) % TK);
            long long gb = boff + (long long)(kb + TK + (t * BV) / TN) * ldb + (n0 + (t * BV) % TN);
            if (A_F32) __builtin_prefetch(&Af[ga], 0, 0); else __builtin_prefetch(&Abf[ga], 0, 0);
            if (B_F32) __builtin_prefetch(&Bf[gb], 0, 0); else __builtin_prefetch(&Bbf[gb], 0, 0);
        }
        __syncthreads();

        // ---- two WMMA K-steps per stage; fragments per ISA 7.12.2 ----
#pragma unroll
        for (int ks = 0; ks < TK; ks += 32) {
            Frag af[FM], bf_[FN];
#pragma unroll
            for (int fm = 0; fm < FM; ++fm) {
                int row = wm * (FM * 16) + fm * 16 + l;
#pragma unroll
                for (int j = 0; j < 8; ++j) {
                    int kk = (j < 4) ? (hi * 8 + 2 * j) : (16 + hi * 8 + 2 * (j - 4));
                    af[fm].u[j] = *(const unsigned int*)&lA[row * TK + ks + kk];
                }
            }
#pragma unroll
            for (int fn = 0; fn < FN; ++fn) {
                int col = wn * (FN * 16) + fn * 16 + l;
#pragma unroll
                for (int j = 0; j < 8; ++j) {
                    int kk = (j < 4) ? (hi * 8 + 2 * j) : (16 + hi * 8 + 2 * (j - 4));
                    bf_[fn].u[j] = *(const unsigned int*)&lB[col * BPAD + ks + kk];
                }
            }
#pragma unroll
            for (int fm = 0; fm < FM; ++fm)
#pragma unroll
                for (int fn = 0; fn < FN; ++fn)
                    acc[fm][fn] = __builtin_amdgcn_wmma_f32_16x16x32_bf16(
                        false, af[fm].v, false, bf_[fn].v, (short)0, acc[fm][fn],
                        false, false);
        }
        __syncthreads();
    }

    // ---- epilogue: C layout = VGPR r -> M = r + 8*hi, N = l ----
    float*          Cf  = (float*)Cp;
    unsigned short* Cbf = (unsigned short*)Cp;
#pragma unroll
    for (int fm = 0; fm < FM; ++fm) {
#pragma unroll
        for (int fn = 0; fn < FN; ++fn) {
#pragma unroll
            for (int r = 0; r < 8; ++r) {
                int row = m0 + wm * (FM * 16) + fm * 16 + hi * 8 + r;
                int col = n0 + wn * (FN * 16) + fn * 16 + l;
                float v = acc[fm][fn][r];
                if (BIAS) v += bias[row];
                long long g = coff + (long long)row * ldc + col;
                if (OUT_F32) Cf[g] = v; else Cbf[g] = f32_to_bf16(v);
            }
        }
    }
}

// ---------------------------------------------------------------------------
// Softmax over n=4096 on each k-row, in-place on bf16 qkv buffer.
// ---------------------------------------------------------------------------
__global__ __launch_bounds__(256) void softmax_k_rows(unsigned short* __restrict__ qkv)
{
    const int bid = blockIdx.x;
    const int b = bid >> 9;          // /512
    const int r = bid & 511;
    unsigned short* row = qkv + ((long long)b * 1536 + 512 + r) * 4096;

    const int t = threadIdx.x;
    float v[16];
    float m = -3.402823466e38f;
#pragma unroll
    for (int i = 0; i < 16; ++i) {
        v[i] = bf16_to_f32(row[i * 256 + t]);
        m = fmaxf(m, v[i]);
    }
    __shared__ float sred[256];
    sred[t] = m; __syncthreads();
    for (int s = 128; s > 0; s >>= 1) { if (t < s) sred[t] = fmaxf(sred[t], sred[t + s]); __syncthreads(); }
    m = sred[0]; __syncthreads();

    float sum = 0.f;
#pragma unroll
    for (int i = 0; i < 16; ++i) { v[i] = __expf(v[i] - m); sum += v[i]; }
    sred[t] = sum; __syncthreads();
    for (int s = 128; s > 0; s >>= 1) { if (t < s) sred[t] += sred[t + s]; __syncthreads(); }
    const float inv = 1.0f / sred[0];
#pragma unroll
    for (int i = 0; i < 16; ++i) row[i * 256 + t] = f32_to_bf16(v[i] * inv);
}

// ---------------------------------------------------------------------------
// context[d,e] = sum_n k_sm[d,n] * v[e,n]   (64x64 per (b,h), K=4096)
// 8 waves split K (512 each, 16 f32 accum frags per wave); LDS reduction.
// Writes context TRANSPOSED: ctxT[e][d], bf16 — feeds step-4 GEMM as A.
// ---------------------------------------------------------------------------
__global__ __launch_bounds__(256) void context_kernel(
    const unsigned short* __restrict__ qkv, unsigned short* __restrict__ ctxT)
{
    __shared__ float red[2][64 * 64];   // 32 KB

    const int z = blockIdx.x;
    const int b = z >> 3, h = z & 7;
    const unsigned short* kbase = qkv + ((long long)b * 1536 + 512  + h * 64) * 4096;
    const unsigned short* vbase = qkv + ((long long)b * 1536 + 1024 + h * 64) * 4096;

    const int t = threadIdx.x;
    const int wave = t >> 5;
    const int lane = t & 31;
    const int l = lane & 15, hi = lane >> 4;

    v8f acc[4][4] = {};
    const int kend = wave * 512 + 512;
    for (int kn = wave * 512; kn < kend; kn += 32) {
        Frag af[4], bfr[4];
#pragma unroll
        for (int mt = 0; mt < 4; ++mt) {
            int rowm = mt * 16 + l;
#pragma unroll
            for (int j = 0; j < 8; ++j) {
                int kk = (j < 4) ? (hi * 8 + 2 * j) : (16 + hi * 8 + 2 * (j - 4));
                af[mt].u[j] = *(const unsigned int*)&kbase[(long long)rowm * 4096 + kn + kk];
            }
        }
#pragma unroll
        for (int et = 0; et < 4; ++et) {
            int cole = et * 16 + l;
#pragma unroll
            for (int j = 0; j < 8; ++j) {
                int kk = (j < 4) ? (hi * 8 + 2 * j) : (16 + hi * 8 + 2 * (j - 4));
                bfr[et].u[j] = *(const unsigned int*)&vbase[(long long)cole * 4096 + kn + kk];
            }
        }
#pragma unroll
        for (int mt = 0; mt < 4; ++mt)
#pragma unroll
            for (int et = 0; et < 4; ++et)
                acc[mt][et] = __builtin_amdgcn_wmma_f32_16x16x32_bf16(
                    false, af[mt].v, false, bfr[et].v, (short)0, acc[mt][et],
                    false, false);
    }

    // cross-wave reduction: 2 LDS regions, 4 phases of wave-pairs
    for (int phase = 0; phase < 4; ++phase) {
        if ((wave >> 1) == phase) {
            float* reg = red[wave & 1];
#pragma unroll
            for (int mt = 0; mt < 4; ++mt)
#pragma unroll
                for (int et = 0; et < 4; ++et)
#pragma unroll
                    for (int r = 0; r < 8; ++r) {
                        int row = mt * 16 + hi * 8 + r;
                        int col = et * 16 + l;
                        if (phase == 0) reg[row * 64 + col]  = acc[mt][et][r];
                        else            reg[row * 64 + col] += acc[mt][et][r];
                    }
        }
        __syncthreads();
    }

    for (int i = t; i < 4096; i += 256) {
        float s = red[0][i] + red[1][i];
        int d = i >> 6, e = i & 63;
        ctxT[(long long)z * 4096 + e * 64 + d] = f32_to_bf16(s);
    }
}

// ---------------------------------------------------------------------------
// Host-side launch
// ---------------------------------------------------------------------------
extern "C" void kernel_launch(void* const* d_in, const int* in_sizes, int n_in,
                              void* d_out, int out_size, void* d_ws, size_t ws_size,
                              hipStream_t stream)
{
    (void)in_sizes; (void)n_in; (void)out_size; (void)ws_size;
    const float* x     = (const float*)d_in[0];   // (16,256,64,64)
    const float* w_qkv = (const float*)d_in[1];   // (1536,256)
    const float* w_out = (const float*)d_in[2];   // (256,512)
    const float* b_out = (const float*)d_in[3];   // (256,)
    float* y = (float*)d_out;                     // (16,256,64,64)

    // workspace (bf16): qkv | ctxT | out
    unsigned short* qkv  = (unsigned short*)d_ws;           // 16*1536*4096
    unsigned short* ctxT = qkv  + 100663296LL;              // 128*64*64
    unsigned short* outb = ctxT + 524288LL;                 // 16*512*4096

    // 1) qkv = w_qkv @ x   : M=1536 K=256 N=4096, per batch
    gemm_bf16_kernel<128, true, true, false, false>
        <<<dim3(32, 12, 16), 256, 0, stream>>>(
            w_qkv, 256, 0LL, 0LL,
            x,     4096, 256LL * 4096, 0LL,
            qkv,   4096, 1536LL * 4096, 0LL,
            nullptr, 256, 1);

    // 2) softmax over n on k rows (in place)
    softmax_k_rows<<<16 * 512, 256, 0, stream>>>(qkv);

    // 3) context^T per (b,h)
    context_kernel<<<128, 256, 0, stream>>>(qkv, ctxT);

    // 4) out[e,n] = sum_d ctxT[e,d] * q[d,n] : M=64 K=64 N=4096, per (b,h)
    gemm_bf16_kernel<64, false, false, false, false>
        <<<dim3(32, 1, 128), 256, 0, stream>>>(
            ctxT, 64,   8LL * 4096,     4096LL,
            qkv,  4096, 1536LL * 4096,  64LL * 4096,   // q region starts at o=0
            outb, 4096, 512LL * 4096,   64LL * 4096,
            nullptr, 64, 8);

    // 5) y = w_out @ out + b_out : M=256 K=512 N=4096, per batch, f32 out
    gemm_bf16_kernel<128, true, false, true, true>
        <<<dim3(32, 2, 16), 256, 0, stream>>>(
            w_out, 512, 0LL, 0LL,
            outb,  4096, 512LL * 4096, 0LL,
            y,     4096, 256LL * 4096, 0LL,
            b_out, 512, 1);
}
